// char_rnn_82463372083776
// MI455X (gfx1250) — compile-verified, matrix operand order checked by reading, not run on hardware
//
#include <hip/hip_runtime.h>

// ---------------------------------------------------------------------------
// Bidirectional char-GRU for MI455X (gfx1250, wave32, WMMA).
//   B=32, S=128, C=32, VOCAB=262, E=128, H=256, 3H=768, N=B*S=4096
// Strategy:
//   * bf16 WMMA 16x16x32 (fp32 accum) for the input projection (K=128) and the
//     recurrent projection (K=256), fused into one time-stepped kernel.
//   * Weights pre-packed once into a lane-contiguous "fragment-ready" bf16
//     layout in d_ws so each B-operand is one 32B global load (L2-hot).
//   * Each wave owns a 32-wide hidden slice across ALL gates, so the GRU gate
//     nonlinearities run entirely in registers in WMMA D-fragment layout.
//   * Empty asm makes the weight pointers opaque per timestep (stops LICM from
//     spilling 576 VGPRs of cached B-fragments to scratch, seen round 1), and
//     explicit address_space(1) casts keep the loads on the GLOBAL path (round
//     2 regressed to FLAT, which ties LOADcnt to DScnt and forces full drains
//     before WMMA).
// ---------------------------------------------------------------------------

#define BDIM 256
#define WPB  32          // words per block
#define Bsz  32
#define Ssz  128
#define Csz  32
#define Esz  128
#define Hsz  256
#define NTOT (Bsz*Ssz)   // 4096

#define AS1 __attribute__((address_space(1)))

typedef __bf16 v16bf __attribute__((ext_vector_type(16)));
typedef __bf16 v2bf  __attribute__((ext_vector_type(2)));
typedef float  v8f   __attribute__((ext_vector_type(8)));
typedef float  f4v   __attribute__((ext_vector_type(4)));

union FragBF { v16bf v; f4v q[2]; };

// ---- global-address-space access helpers (force global_load/global_store) --
__device__ __forceinline__ v16bf g_load_v16bf(const __bf16* p) {
  return *(const AS1 v16bf*)(uintptr_t)p;
}
__device__ __forceinline__ f4v g_load_f4v(const float* p) {
  return *(const AS1 f4v*)(uintptr_t)p;
}
__device__ __forceinline__ int g_load_i32(const int* p) {
  return *(const AS1 int*)(uintptr_t)p;
}
__device__ __forceinline__ float g_load_f32(const float* p) {
  return *(const AS1 float*)(uintptr_t)p;
}
__device__ __forceinline__ void g_store_f32(float* p, float v) {
  *(AS1 float*)(uintptr_t)p = v;
}
__device__ __forceinline__ void g_store_bf16(__bf16* p, __bf16 v) {
  *(AS1 __bf16*)(uintptr_t)p = v;
}

__device__ __forceinline__ float sigmoidf_(float x) {
  return 1.0f / (1.0f + __expf(-x));
}

__device__ __forceinline__ v8f v8f_zero() {
  v8f r;
#pragma unroll
  for (int i = 0; i < 8; ++i) r[i] = 0.0f;
  return r;
}

// K index of element e within a 16-element bf16 WMMA A/B operand (ISA 7.12.2):
// lanes 0-15 (hlf=0): VGPR0..3 -> K 0..7, VGPR4..7 -> K 16..23
// lanes 16-31 (hlf=1): VGPR0..3 -> K 8..15, VGPR4..7 -> K 24..31
__device__ __forceinline__ int wmma_k_of(int e, int hlf) {
  int j = e >> 1, b = e & 1;
  int base = (j < 4) ? (2 * j) : (16 + 2 * (j - 4));
  return base + 8 * hlf + b;
}

// ---------------------------------------------------------------------------
// Pack W_ih (3H x E) and W_hh (3H x H) for both directions into bf16
// fragment-ready layout:
//   wih_pk[((dir*4 + ks)*48 + tile)*32 + lane][e]   (e = 0..15)
//   whh_pk starts at element offset 2*4*48*512
// B[k][g] = W[g][k] with k = ks*32 + wmma_k_of(e, lane>>4), g = tile*16+lane%16
// ---------------------------------------------------------------------------
__global__ __launch_bounds__(BDIM)
void pack_weights_bf16(const float* __restrict__ Wih_fw,
                       const float* __restrict__ Whh_fw,
                       const float* __restrict__ Wih_bw,
                       const float* __restrict__ Whh_bw,
                       __bf16* __restrict__ wpk) {
  int t = blockIdx.x * BDIM + threadIdx.x;
  const int IH_TASKS = 2 * 4 * 48 * 32;   // 12288
  const int HH_TASKS = 2 * 8 * 48 * 32;   // 24576
  if (t < IH_TASKS) {
    int lane = t & 31;
    int tile = (t >> 5) % 48;
    int ks   = ((t >> 5) / 48) % 4;
    int dir  = (t >> 5) / (48 * 4);
    const float* W = dir ? Wih_bw : Wih_fw;  // [768 x 128]
    int g   = tile * 16 + (lane & 15);
    int hlf = lane >> 4;
    __bf16* dst = wpk + (size_t)t * 16;
#pragma unroll
    for (int e = 0; e < 16; ++e) {
      int k = ks * 32 + wmma_k_of(e, hlf);
      g_store_bf16(dst + e, (__bf16)g_load_f32(W + g * Esz + k));
    }
  } else if (t < IH_TASKS + HH_TASKS) {
    int u    = t - IH_TASKS;
    int lane = u & 31;
    int tile = (u >> 5) % 48;
    int ks   = ((u >> 5) / 48) % 8;
    int dir  = (u >> 5) / (48 * 8);
    const float* W = dir ? Whh_bw : Whh_fw;  // [768 x 256]
    int g   = tile * 16 + (lane & 15);
    int hlf = lane >> 4;
    __bf16* dst = wpk + (size_t)IH_TASKS * 16 + (size_t)u * 16;
#pragma unroll
    for (int e = 0; e < 16; ++e) {
      int k = ks * 32 + wmma_k_of(e, hlf);
      g_store_bf16(dst + e, (__bf16)g_load_f32(W + g * Hsz + k));
    }
  }
}

// ---------------------------------------------------------------------------
// Fused GRU kernel: grid = (N/WPB, 2 directions), block = 256 threads (8 waves)
// ---------------------------------------------------------------------------
__global__ __launch_bounds__(BDIM)
void char_gru_dir_kernel(const int*   __restrict__ chars,       // [N, C]
                         const int*   __restrict__ chars_mask,  // [N, C]
                         const int*   __restrict__ data_mask,   // [N]
                         const float* __restrict__ embed,       // [VOCAB, E]
                         const float* __restrict__ b_ih_fw,
                         const float* __restrict__ b_hh_fw,
                         const float* __restrict__ b_ih_bw,
                         const float* __restrict__ b_hh_bw,
                         const __bf16* __restrict__ wpk,
                         float* __restrict__ out)               // [N, 2H]
{
  __shared__ __align__(16) __bf16 sXA[WPB * Esz];   // 8 KB  (A operand, x step)
  __shared__ __align__(16) __bf16 sHA[WPB * Hsz];   // 16 KB (A operand, h)
  __shared__ float sMask[WPB];

  const int tid  = threadIdx.x;
  const int lane = tid & 31;
  const int wave = tid >> 5;       // 0..7, owns hidden cols [wave*32, wave*32+32)
  const int col  = lane & 15;      // N / M-in-lane index for WMMA fragments
  const int hlf  = lane >> 4;
  const int dir  = blockIdx.y;     // 0 = forward, 1 = backward
  const int w0   = blockIdx.x * WPB;

  const float* bih = dir ? b_ih_bw : b_ih_fw;
  const float* bhh = dir ? b_hh_bw : b_hh_fw;

  const __bf16* wih_p = wpk + (size_t)(dir * 4) * 48 * 512;
  const __bf16* whh_p = wpk + (size_t)2 * 4 * 48 * 512 + (size_t)(dir * 8) * 48 * 512;

  // Per-column biases for this wave's hidden slice (constant over time).
  float bR[2], bZ[2], bXN[2], bHN[2];
#pragma unroll
  for (int tt = 0; tt < 2; ++tt) {
    int jj = wave * 32 + tt * 16 + col;
    bR[tt]  = g_load_f32(bih + jj)           + g_load_f32(bhh + jj);
    bZ[tt]  = g_load_f32(bih + Hsz + jj)     + g_load_f32(bhh + Hsz + jj);
    bXN[tt] = g_load_f32(bih + 2 * Hsz + jj);
    bHN[tt] = g_load_f32(bhh + 2 * Hsz + jj);
  }

  // h0 = 0 (LDS copy + register copy in D-fragment layout)
  for (int i = tid; i < WPB * Hsz; i += BDIM) sHA[i] = (__bf16)0.0f;
  v8f hreg[2][2];
#pragma unroll
  for (int mt = 0; mt < 2; ++mt)
#pragma unroll
    for (int tt = 0; tt < 2; ++tt) hreg[mt][tt] = v8f_zero();
  __syncthreads();

#pragma unroll 1
  for (int c = 0; c < Csz; ++c) {
    const int cc = dir ? (Csz - 1 - c) : c;

    // Opaque weight pointers: forbid LICM from caching 72 B-fragments (576
    // VGPRs) across timesteps and spilling to scratch. Zero instructions.
    asm volatile("" : "+s"(wih_p), "+s"(whh_p));

    // ---- phase 1: gather embedding rows (fp32 -> bf16) into sXA ----
    {
      int word = tid >> 3, seg = tid & 7;       // 8 segs of 16 cols per word
      int n  = w0 + word;
      int ch = g_load_i32(chars + n * Csz + cc);
      if (seg == 0) sMask[word] = (float)g_load_i32(chars_mask + n * Csz + cc);
      const float* src = embed + (size_t)ch * Esz + seg * 16;
      f4v r0 = g_load_f4v(src);
      f4v r1 = g_load_f4v(src + 4);
      f4v r2 = g_load_f4v(src + 8);
      f4v r3 = g_load_f4v(src + 12);
      v2bf* dst = (v2bf*)&sXA[word * Esz + seg * 16];
#pragma unroll
      for (int i = 0; i < 2; ++i) { v2bf p; p[0] = (__bf16)r0[2*i]; p[1] = (__bf16)r0[2*i+1]; dst[i]     = p; }
#pragma unroll
      for (int i = 0; i < 2; ++i) { v2bf p; p[0] = (__bf16)r1[2*i]; p[1] = (__bf16)r1[2*i+1]; dst[2 + i] = p; }
#pragma unroll
      for (int i = 0; i < 2; ++i) { v2bf p; p[0] = (__bf16)r2[2*i]; p[1] = (__bf16)r2[2*i+1]; dst[4 + i] = p; }
#pragma unroll
      for (int i = 0; i < 2; ++i) { v2bf p; p[0] = (__bf16)r3[2*i]; p[1] = (__bf16)r3[2*i+1]; dst[6 + i] = p; }
    }
    __syncthreads();

    // ---- phase 2: GEMMs. acc tiles: [0,1]=r, [2,3]=z, [4,5]=x·n, [6,7]=h·n
    v8f acc[2][8];
#pragma unroll
    for (int mt = 0; mt < 2; ++mt)
#pragma unroll
      for (int i = 0; i < 8; ++i) acc[mt][i] = v8f_zero();

    // input projection: A = sXA (32x128), B = W_ih^T  (4 k-steps of 32)
#pragma unroll
    for (int ks = 0; ks < 4; ++ks) {
      FragBF a[2];
#pragma unroll
      for (int mt = 0; mt < 2; ++mt) {
        int row = mt * 16 + col;
        int kb  = ks * 32 + 8 * hlf;
        a[mt].q[0] = *(const f4v*)&sXA[row * Esz + kb];
        a[mt].q[1] = *(const f4v*)&sXA[row * Esz + kb + 16];
      }
#pragma unroll
      for (int ti = 0; ti < 6; ++ti) {
        int region = ti >> 1;                          // 0=r, 1=z, 2=n
        int gtile  = region * 16 + (wave << 1) + (ti & 1);
        v16bf b = g_load_v16bf(wih_p + ((size_t)ks * 48 + gtile) * 512 + lane * 16);
#pragma unroll
        for (int mt = 0; mt < 2; ++mt)
          acc[mt][ti] = __builtin_amdgcn_wmma_f32_16x16x32_bf16(
              false, a[mt].v, false, b, (short)0, acc[mt][ti], false, false);
      }
    }

    // recurrent projection: A = sHA (32x256), B = W_hh^T  (8 k-steps of 32)
#pragma unroll
    for (int ks = 0; ks < 8; ++ks) {
      FragBF a[2];
#pragma unroll
      for (int mt = 0; mt < 2; ++mt) {
        int row = mt * 16 + col;
        int kb  = ks * 32 + 8 * hlf;
        a[mt].q[0] = *(const f4v*)&sHA[row * Hsz + kb];
        a[mt].q[1] = *(const f4v*)&sHA[row * Hsz + kb + 16];
      }
#pragma unroll
      for (int ti = 0; ti < 6; ++ti) {
        int region = ti >> 1;
        int gtile  = region * 16 + (wave << 1) + (ti & 1);
        int ai     = (ti < 4) ? ti : (ti + 2);        // h·n goes to tiles 6,7
        v16bf b = g_load_v16bf(whh_p + ((size_t)ks * 48 + gtile) * 512 + lane * 16);
#pragma unroll
        for (int mt = 0; mt < 2; ++mt)
          acc[mt][ai] = __builtin_amdgcn_wmma_f32_16x16x32_bf16(
              false, a[mt].v, false, b, (short)0, acc[mt][ai], false, false);
      }
    }

    // ---- gate math, entirely in registers (fp32) ----
#pragma unroll
    for (int mt = 0; mt < 2; ++mt) {
#pragma unroll
      for (int tt = 0; tt < 2; ++tt) {
#pragma unroll
        for (int e = 0; e < 8; ++e) {
          float rv = sigmoidf_(acc[mt][tt][e]     + bR[tt]);
          float zv = sigmoidf_(acc[mt][2 + tt][e] + bZ[tt]);
          float nv = tanhf(acc[mt][4 + tt][e] + bXN[tt] +
                           rv * (acc[mt][6 + tt][e] + bHN[tt]));
          float ho = hreg[mt][tt][e];
          float hn = (1.0f - zv) * nv + zv * ho;
          float m  = sMask[mt * 16 + e + 8 * hlf];
          hreg[mt][tt][e] = m * hn + (1.0f - m) * ho;
        }
      }
    }
    __syncthreads();   // all waves done reading sHA/sXA before republishing h

    // ---- phase 3: publish new h (bf16) for the next step's A operand ----
#pragma unroll
    for (int mt = 0; mt < 2; ++mt)
#pragma unroll
      for (int tt = 0; tt < 2; ++tt)
#pragma unroll
        for (int e = 0; e < 8; ++e) {
          int row = mt * 16 + e + 8 * hlf;
          sHA[row * Hsz + wave * 32 + tt * 16 + col] = (__bf16)hreg[mt][tt][e];
        }
    __syncthreads();
  }

  // ---- write output: out[n, dir*H + j] = h_final * data_mask[n] ----
#pragma unroll
  for (int mt = 0; mt < 2; ++mt)
#pragma unroll
    for (int tt = 0; tt < 2; ++tt)
#pragma unroll
      for (int e = 0; e < 8; ++e) {
        int row = mt * 16 + e + 8 * hlf;
        int n   = w0 + row;
        float dm = (float)g_load_i32(data_mask + n);
        g_store_f32(out + (size_t)n * (2 * Hsz) + dir * Hsz + wave * 32 + tt * 16 + col,
                    hreg[mt][tt][e] * dm);
      }
}

// ---------------------------------------------------------------------------
extern "C" void kernel_launch(void* const* d_in, const int* in_sizes, int n_in,
                              void* d_out, int out_size, void* d_ws, size_t ws_size,
                              hipStream_t stream) {
  (void)in_sizes; (void)n_in; (void)out_size; (void)ws_size;
  const int*   chars      = (const int*)  d_in[0];
  const int*   chars_mask = (const int*)  d_in[1];
  const int*   data_mask  = (const int*)  d_in[2];
  const float* embed      = (const float*)d_in[3];
  const float* W_ih_fw    = (const float*)d_in[4];
  const float* W_hh_fw    = (const float*)d_in[5];
  const float* b_ih_fw    = (const float*)d_in[6];
  const float* b_hh_fw    = (const float*)d_in[7];
  const float* W_ih_bw    = (const float*)d_in[8];
  const float* W_hh_bw    = (const float*)d_in[9];
  const float* b_ih_bw    = (const float*)d_in[10];
  const float* b_hh_bw    = (const float*)d_in[11];
  float*  out = (float*)d_out;
  __bf16* wpk = (__bf16*)d_ws;   // needs 2*(4+8)*48*32*16*2B = 1,179,648 bytes

  // Pack weights into fragment-ready bf16 layout (36864 lane-tasks).
  pack_weights_bf16<<<144, BDIM, 0, stream>>>(W_ih_fw, W_hh_fw, W_ih_bw, W_hh_bw, wpk);

  // 128 word-tiles x 2 directions, 8 waves each.
  char_gru_dir_kernel<<<dim3(NTOT / WPB, 2, 1), BDIM, 0, stream>>>(
      chars, chars_mask, data_mask, embed,
      b_ih_fw, b_hh_fw, b_ih_bw, b_hh_bw, wpk, out);
}
